// GCNModel_68143951118910
// MI455X (gfx1250) — matrix operand order repaired and checked by reading, not compiled, not run on hardware
//
#include <hip/hip_runtime.h>
#include <hip/hip_bf16.h>
#include <stddef.h>

typedef __attribute__((ext_vector_type(16))) __bf16 v16bf;
typedef __attribute__((ext_vector_type(8)))  __bf16 v8bf;
typedef __attribute__((ext_vector_type(4)))  __bf16 v4bf;
typedef __attribute__((ext_vector_type(8)))  float  v8f;
typedef int v4i_g __attribute__((vector_size(16)));   // matches builtin's param type

#define N_ROW   8192
#define FDIM    64
#define CDIM    10
#define KSPLIT  8
#define MTILE   128
#define KTILE   64
#define LDSA    72      // padded bf16 row stride (36 dwords -> conflict-free 16B chunks)
#define LDSB    72
#define XPAD    65

#define AS1 __attribute__((address_space(1)))
#define AS3 __attribute__((address_space(3)))

#if __has_builtin(__builtin_amdgcn_global_load_async_to_lds_b128)
#define HAVE_ASYNC_LDS 1
#else
#define HAVE_ASYNC_LDS 0
#endif

static __device__ __forceinline__ void wait_async_zero() {
#if __has_builtin(__builtin_amdgcn_s_wait_asynccnt)
    __builtin_amdgcn_s_wait_asynccnt(0);
#else
    asm volatile("s_wait_asynccnt 0" ::: "memory");
#endif
}

// ---------------------------------------------------------------------------
// Kernel 1: s = rowsum(A); d = s > 0 ? rsqrt(s) : 0
// ---------------------------------------------------------------------------
__global__ __launch_bounds__(256) void rowsum_rsqrt_kernel(
    const float* __restrict__ A, float* __restrict__ dvec)
{
    const int lane = threadIdx.x & 31;
    const int wv   = threadIdx.x >> 5;
    const int row  = blockIdx.x * 8 + wv;
    const float* ar = A + (size_t)row * N_ROW;
    float s = 0.f;
    #pragma unroll 4
    for (int j = 0; j < 64; ++j) {
        const float4 v = *(const float4*)(ar + j * 128 + lane * 4);
        s += v.x + v.y + v.z + v.w;
    }
    #pragma unroll
    for (int off = 16; off > 0; off >>= 1) s += __shfl_xor(s, off, 32);
    if (lane == 0) dvec[row] = (s > 0.f) ? rsqrtf(s) : 0.f;
}

// ---------------------------------------------------------------------------
// Kernel 2: yT[c][r] = bf16( d[r] * (X @ W)[r][c] )   (X fp32 [8192][64])
// ---------------------------------------------------------------------------
__global__ __launch_bounds__(256) void dense_scale_yT_kernel(
    const float* __restrict__ X, const float* __restrict__ W,
    const float* __restrict__ dvec, __bf16* __restrict__ yT)
{
    __shared__ float Xs[64 * XPAD];
    __shared__ float Ws[64 * 64];
    const int tid   = threadIdx.x;
    const int rbase = blockIdx.x * 64;

    #pragma unroll
    for (int i = 0; i < 16; ++i) Ws[i * 256 + tid] = W[i * 256 + tid];
    #pragma unroll
    for (int i = 0; i < 4; ++i) {
        const int idx = i * 256 + tid;            // float4 index 0..1023
        const int fr = idx >> 4, fc = idx & 15;
        const float4 v = *(const float4*)(X + (size_t)(rbase + fr) * FDIM + fc * 4);
        Xs[fr * XPAD + fc * 4 + 0] = v.x;
        Xs[fr * XPAD + fc * 4 + 1] = v.y;
        Xs[fr * XPAD + fc * 4 + 2] = v.z;
        Xs[fr * XPAD + fc * 4 + 3] = v.w;
    }
    __syncthreads();

    const int r  = tid & 63;
    const int cg = tid >> 6;
    const float dr = dvec[rbase + r];
    #pragma unroll 4
    for (int cc = 0; cc < 16; ++cc) {
        const int c = cg * 16 + cc;
        float acc = 0.f;
        #pragma unroll 8
        for (int k = 0; k < 64; ++k) acc += Xs[r * XPAD + k] * Ws[k * 64 + c];
        yT[(size_t)c * N_ROW + rbase + r] = (__bf16)(dr * acc);
    }
}

// ---------------------------------------------------------------------------
// Kernel 3: partial[ks] = A[:, kslice] @ Y[kslice, :]  via bf16 WMMA.
// Software-pipelined: next A tile staged in regs, next B tile copied
// global->LDS with async-to-LDS (double-buffered) while WMMAs run.
// ---------------------------------------------------------------------------
__global__ __launch_bounds__(256) void spmm_wmma_kernel(
    const float* __restrict__ A, const __bf16* __restrict__ yT,
    float* __restrict__ partial)
{
    __shared__ __bf16 As[MTILE * LDSA];       // 18 KB
    __shared__ __bf16 Bs[2][FDIM * LDSB];     // 2 x 9 KB

    const int tid  = threadIdx.x;
    const int lane = tid & 31;
    const int wv   = tid >> 5;
    const int half = lane >> 4;
    const int l16  = lane & 15;

    const int mbase  = blockIdx.x * MTILE;
    const int kslice = blockIdx.y;
    const int k0     = kslice * (N_ROW / KSPLIT);
    const int NITER  = (N_ROW / KSPLIT) / KTILE;   // 16

    v8f acc[4];
    #pragma unroll
    for (int t = 0; t < 4; ++t)
        #pragma unroll
        for (int i = 0; i < 8; ++i) acc[t][i] = 0.f;

    const int ch = tid & 15;   // A-tile float4 column chunk
    const int r0 = tid >> 4;   // A-tile base row (0..15)
    const int bn = tid >> 2;   // B-tile row (n, 0..63)
    const int sg = tid & 3;    // B-tile 16-elem segment

    const float*  aptr = A  + (size_t)(mbase + r0) * N_ROW + k0 + ch * 4;
    const __bf16* bptr = yT + (size_t)bn * N_ROW + k0 + sg * 16;
    const int bofs = bn * LDSB + sg * 16;

    float4 areg[8];
#if !HAVE_ASYNC_LDS
    v8bf breg0, breg1;
#endif

    // ---- prologue: stage tile 0 ----
    #pragma unroll
    for (int p = 0; p < 8; ++p)
        areg[p] = *(const float4*)(aptr + (size_t)p * 16 * N_ROW);
#if HAVE_ASYNC_LDS
    __builtin_amdgcn_global_load_async_to_lds_b128(
        (AS1 v4i_g*)bptr,       (AS3 v4i_g*)&Bs[0][bofs],     0, 0);
    __builtin_amdgcn_global_load_async_to_lds_b128(
        (AS1 v4i_g*)(bptr + 8), (AS3 v4i_g*)&Bs[0][bofs + 8], 0, 0);
#else
    breg0 = *(const v8bf*)bptr;
    breg1 = *(const v8bf*)(bptr + 8);
#endif

    int cur = 0;
    for (int it = 0; it < NITER; ++it) {
        // ---- commit staged tile it into LDS ----
        #pragma unroll
        for (int p = 0; p < 8; ++p) {
            v4bf b;
            b[0] = (__bf16)areg[p].x; b[1] = (__bf16)areg[p].y;
            b[2] = (__bf16)areg[p].z; b[3] = (__bf16)areg[p].w;
            *(v4bf*)(&As[(r0 + p * 16) * LDSA + ch * 4]) = b;
        }
#if HAVE_ASYNC_LDS
        wait_async_zero();                    // B tile it landed in Bs[cur]
#else
        *(v8bf*)(&Bs[cur][bofs])     = breg0;
        *(v8bf*)(&Bs[cur][bofs + 8]) = breg1;
#endif
        __syncthreads();

        // ---- issue tile it+1 (overlaps with WMMA below) ----
        if (it + 1 < NITER) {
            const float*  ap = aptr + (it + 1) * KTILE;
            #pragma unroll
            for (int p = 0; p < 8; ++p)
                areg[p] = *(const float4*)(ap + (size_t)p * 16 * N_ROW);
            const __bf16* bp = bptr + (it + 1) * KTILE;
#if HAVE_ASYNC_LDS
            __builtin_amdgcn_global_load_async_to_lds_b128(
                (AS1 v4i_g*)bp,       (AS3 v4i_g*)&Bs[cur ^ 1][bofs],     0, 0);
            __builtin_amdgcn_global_load_async_to_lds_b128(
                (AS1 v4i_g*)(bp + 8), (AS3 v4i_g*)&Bs[cur ^ 1][bofs + 8], 0, 0);
#else
            breg0 = *(const v8bf*)bp;
            breg1 = *(const v8bf*)(bp + 8);
#endif
        }

        // ---- compute: 2 K-steps x 4 N-tiles of 16x16x32 bf16 WMMA ----
        const int ra = wv * 16 + l16;
        #pragma unroll
        for (int ks = 0; ks < 2; ++ks) {
            const int koff = ks * 32;
            const v8bf alo = *(const v8bf*)(&As[ra * LDSA + koff + half * 16]);
            const v8bf ahi = *(const v8bf*)(&As[ra * LDSA + koff + half * 16 + 8]);
            const v16bf afrag = __builtin_shufflevector(alo, ahi,
                0,1,2,3,4,5,6,7,8,9,10,11,12,13,14,15);
            #pragma unroll
            for (int nt = 0; nt < 4; ++nt) {
                const int n = nt * 16 + l16;
                const v8bf blo = *(const v8bf*)(&Bs[cur][n * LDSB + koff + half * 16]);
                const v8bf bhi = *(const v8bf*)(&Bs[cur][n * LDSB + koff + half * 16 + 8]);
                const v16bf bfrag = __builtin_shufflevector(blo, bhi,
                    0,1,2,3,4,5,6,7,8,9,10,11,12,13,14,15);
                acc[nt] = __builtin_amdgcn_wmma_f32_16x16x32_bf16(
                    false, afrag, false, bfrag, (short)0, acc[nt], false, false);
            }
        }
        __syncthreads();
        cur ^= 1;
    }

    // ---- write split-K partials (C/D layout: VGPR v -> M = v + 8*half, N = l16)
    float* pout = partial + ((size_t)kslice * N_ROW + mbase) * FDIM;
    #pragma unroll
    for (int nt = 0; nt < 4; ++nt) {
        #pragma unroll
        for (int v = 0; v < 8; ++v) {
            const int row = wv * 16 + half * 8 + v;
            const int col = nt * 16 + l16;
            pout[(size_t)row * FDIM + col] = acc[nt][v];
        }
    }
}

// ---------------------------------------------------------------------------
// Kernel 4: x = relu(d * sum_ks partial);  yT = bf16(d * (x @ W))
// ---------------------------------------------------------------------------
__global__ __launch_bounds__(256) void reduce_relu_dense_yT_kernel(
    const float* __restrict__ partial, const float* __restrict__ dvec,
    const float* __restrict__ W, __bf16* __restrict__ yT)
{
    __shared__ float Xs[64 * XPAD];
    __shared__ float Ws[64 * 64];
    const int tid   = threadIdx.x;
    const int rbase = blockIdx.x * 64;

    #pragma unroll
    for (int i = 0; i < 16; ++i) Ws[i * 256 + tid] = W[i * 256 + tid];
    #pragma unroll
    for (int i = 0; i < 16; ++i) {
        const int flat = i * 256 + tid;           // 0..4095
        const int r = flat >> 6, c = flat & 63;
        float s = 0.f;
        #pragma unroll
        for (int ks = 0; ks < KSPLIT; ++ks)
            s += partial[((size_t)ks * N_ROW + rbase + r) * FDIM + c];
        const float v = dvec[rbase + r] * s;
        Xs[r * XPAD + c] = v > 0.f ? v : 0.f;
    }
    __syncthreads();

    const int r  = tid & 63;
    const int cg = tid >> 6;
    const float dr = dvec[rbase + r];
    #pragma unroll 4
    for (int cc = 0; cc < 16; ++cc) {
        const int c = cg * 16 + cc;
        float acc = 0.f;
        #pragma unroll 8
        for (int k = 0; k < 64; ++k) acc += Xs[r * XPAD + k] * Ws[k * 64 + c];
        yT[(size_t)c * N_ROW + rbase + r] = (__bf16)(dr * acc);
    }
}

// ---------------------------------------------------------------------------
// Kernel 5: x = relu(d * sum_ks partial);  out = x @ Wc + bc   (fp32 logits)
// ---------------------------------------------------------------------------
__global__ __launch_bounds__(256) void reduce_relu_logits_kernel(
    const float* __restrict__ partial, const float* __restrict__ dvec,
    const float* __restrict__ Wc, const float* __restrict__ bc,
    float* __restrict__ out)
{
    __shared__ float Xs[64 * XPAD];
    __shared__ float Wcs[64 * CDIM];
    __shared__ float bcs[CDIM];
    const int tid   = threadIdx.x;
    const int rbase = blockIdx.x * 64;

    for (int i = tid; i < 64 * CDIM; i += 256) Wcs[i] = Wc[i];
    if (tid < CDIM) bcs[tid] = bc[tid];
    #pragma unroll
    for (int i = 0; i < 16; ++i) {
        const int flat = i * 256 + tid;
        const int r = flat >> 6, c = flat & 63;
        float s = 0.f;
        #pragma unroll
        for (int ks = 0; ks < KSPLIT; ++ks)
            s += partial[((size_t)ks * N_ROW + rbase + r) * FDIM + c];
        const float v = dvec[rbase + r] * s;
        Xs[r * XPAD + c] = v > 0.f ? v : 0.f;
    }
    __syncthreads();

    for (int o = tid; o < 64 * CDIM; o += 256) {
        const int r = o / CDIM, c = o % CDIM;
        float acc = bcs[c];
        #pragma unroll 8
        for (int k = 0; k < 64; ++k) acc += Xs[r * XPAD + k] * Wcs[k * CDIM + c];
        out[(size_t)(rbase + r) * CDIM + c] = acc;
    }
}

// ---------------------------------------------------------------------------
extern "C" void kernel_launch(void* const* d_in, const int* in_sizes, int n_in,
                              void* d_out, int out_size, void* d_ws, size_t ws_size,
                              hipStream_t stream)
{
    (void)in_sizes; (void)n_in; (void)out_size; (void)ws_size;
    const float* A   = (const float*)d_in[0];
    const float* emb = (const float*)d_in[1];
    const float* W1  = (const float*)d_in[2];
    const float* W2  = (const float*)d_in[3];
    const float* Wc  = (const float*)d_in[4];
    const float* bc  = (const float*)d_in[5];
    float* out = (float*)d_out;

    char* ws = (char*)d_ws;
    float*  dvec    = (float*)ws;                                        // 32 KB
    __bf16* yT      = (__bf16*)(ws + 32768);                             // 1 MB
    float*  partial = (float*)(ws + 32768 + (size_t)FDIM * N_ROW * 2);   // 16 MB

    rowsum_rsqrt_kernel<<<N_ROW / 8, 256, 0, stream>>>(A, dvec);
    dense_scale_yT_kernel<<<N_ROW / 64, 256, 0, stream>>>(emb, W1, dvec, yT);
    spmm_wmma_kernel<<<dim3(N_ROW / MTILE, KSPLIT), 256, 0, stream>>>(A, yT, partial);
    reduce_relu_dense_yT_kernel<<<N_ROW / 64, 256, 0, stream>>>(partial, dvec, W2, yT);
    spmm_wmma_kernel<<<dim3(N_ROW / MTILE, KSPLIT), 256, 0, stream>>>(A, yT, partial);
    reduce_relu_logits_kernel<<<N_ROW / 64, 256, 0, stream>>>(partial, dvec, Wc, bc, out);
}
